// Pointnet2BackboneDecoder_8134668059281
// MI455X (gfx1250) — matrix operand order repaired and checked by reading, not compiled, not run on hardware
//
#include <hip/hip_runtime.h>

typedef unsigned short u16;
typedef __bf16 bf16_t;
typedef bf16_t v16bf __attribute__((ext_vector_type(16)));
typedef float  v8f   __attribute__((ext_vector_type(8)));
typedef unsigned int v8u __attribute__((ext_vector_type(8)));

#define BATCH 8
#define NPTS  20000

__device__ __forceinline__ u16 f2bf(float f) {
    unsigned u = __float_as_uint(f);
    u += 0x7FFFu + ((u >> 16) & 1u);          // round-to-nearest-even
    return (u16)(u >> 16);
}
__device__ __forceinline__ float bf2f(u16 h) {
    return __uint_as_float(((unsigned)h) << 16);
}

// ---------------------------------------------------------------- split ----
__global__ void k_split(const float* __restrict__ pc, float* __restrict__ xyz,
                        float* __restrict__ f, int total) {
    int i = blockIdx.x * blockDim.x + threadIdx.x;
    if (i >= total) return;
    const float* p = pc + (size_t)i * 4;
    xyz[(size_t)i*3+0] = p[0]; xyz[(size_t)i*3+1] = p[1]; xyz[(size_t)i*3+2] = p[2];
    f[i] = p[3];
}

// ----------------------------------------------------------------- FPS -----
// one block per batch; running min-dist array in global scratch (L2 resident)
__global__ void k_fps(const float* __restrict__ xyz, float* __restrict__ dists,
                      int n, int npoint, int* __restrict__ fi) {
    int b = blockIdx.x, tid = threadIdx.x, bd = blockDim.x;
    const float* x = xyz + (size_t)b * n * 3;
    float* dd = dists + (size_t)b * n;
    __shared__ float redv[256];
    __shared__ int   redi[256];
    __shared__ float lastp[3];
    for (int j = tid; j < n; j += bd) dd[j] = 1e10f;
    if (tid == 0) {
        fi[(size_t)b * npoint] = 0;
        lastp[0] = x[0]; lastp[1] = x[1]; lastp[2] = x[2];
    }
    __syncthreads();
    for (int it = 1; it < npoint; ++it) {
        float lx = lastp[0], ly = lastp[1], lz = lastp[2];
        float best = -1.f; int bi = 0;
        for (int j = tid; j < n; j += bd) {
            float dx = x[3*j] - lx, dy = x[3*j+1] - ly, dz = x[3*j+2] - lz;
            float d = dx*dx + dy*dy + dz*dz;
            float dm = dd[j]; dm = d < dm ? d : dm; dd[j] = dm;
            if (dm > best) { best = dm; bi = j; }
        }
        redv[tid] = best; redi[tid] = bi;
        __syncthreads();
        for (int s = bd >> 1; s > 0; s >>= 1) {
            if (tid < s) {
                float ov = redv[tid+s]; int oi = redi[tid+s];
                if (ov > redv[tid] || (ov == redv[tid] && oi < redi[tid])) {
                    redv[tid] = ov; redi[tid] = oi;
                }
            }
            __syncthreads();
        }
        if (tid == 0) {
            int nx = redi[0];
            fi[(size_t)b * npoint + it] = nx;
            lastp[0] = x[3*nx]; lastp[1] = x[3*nx+1]; lastp[2] = x[3*nx+2];
        }
        __syncthreads();
    }
}

// ---------------------------------------------------------- gather xyz -----
__global__ void k_gather_xyz(const float* __restrict__ xyz, const int* __restrict__ fi,
                             float* __restrict__ nxyz, int n, int np, int total) {
    int t = blockIdx.x * blockDim.x + threadIdx.x;
    if (t >= total) return;
    int b = t / np; int j = fi[t];
    const float* p = xyz + ((size_t)b * n + j) * 3;
    nxyz[(size_t)t*3+0] = p[0]; nxyz[(size_t)t*3+1] = p[1]; nxyz[(size_t)t*3+2] = p[2];
}

// ---------------------------------------------------------- ball query -----
// one wave per query; ordered in-radius compaction via ballot, fill with first
__global__ __launch_bounds__(128)
void k_ballq(const float* __restrict__ nxyz, const float* __restrict__ xyz,
             int n, int np, int ns, float r2, int* __restrict__ idx, int totalQ) {
    __shared__ int sbuf[4][32];
    int wv = threadIdx.x >> 5;
    int lane = threadIdx.x & 31;
    int q = blockIdx.x * 4 + wv;
    if (q >= totalQ) return;
    volatile int* wsq = sbuf[wv];
    int b = q / np;
    const float* pq = nxyz + (size_t)q * 3;
    float qx = pq[0], qy = pq[1], qz = pq[2];
    const float* px = xyz + (size_t)b * n * 3;
    int count = 0;
    for (int base = 0; base < n && count < ns; base += 32) {
        int j = base + lane;
        bool in = false;
        if (j < n) {
            float dx = px[3*j] - qx, dy = px[3*j+1] - qy, dz = px[3*j+2] - qz;
            in = (dx*dx + dy*dy + dz*dz) < r2;
        }
        unsigned m = (unsigned)__ballot(in);
        int pre = __popc(m & ((1u << lane) - 1u));
        if (in) { int pos = count + pre; if (pos < ns) wsq[pos] = j; }
        count += __popc(m);
    }
    int c = count < ns ? count : ns;
    int* o = idx + (size_t)q * ns;
    if (lane < ns) o[lane] = (lane < c) ? wsq[lane] : wsq[0];
}

// ------------------------------------------- SA group: build bf16 A rows ---
__global__ void k_build_sa(const float* __restrict__ xyz, const float* __restrict__ nxyz,
                           const float* __restrict__ feats, const int* __restrict__ idx,
                           u16* __restrict__ A, int n, int np, int ns, int Cf, int Kp,
                           float invr, int b) {
    int r = blockIdx.x;                 // 0 .. np*ns-1 (per batch)
    int s = r / ns, kk = r % ns;
    int j = idx[((size_t)b * np + s) * ns + kk];
    const float* pj = xyz  + ((size_t)b * n  + j) * 3;
    const float* pc = nxyz + ((size_t)b * np + s) * 3;
    const float* fj = feats + ((size_t)b * n + j) * Cf;
    u16* Ar = A + (size_t)r * Kp;
    for (int c = threadIdx.x; c < Kp; c += blockDim.x) {
        float v;
        if (c < 3)           v = (pj[c] - pc[c]) * invr;
        else if (c < 3 + Cf) v = fj[c - 3];
        else                 v = 0.f;
        Ar[c] = f2bf(v);
    }
}

// ------------------------------------------------- weight convert+pack -----
// w: [cin][cout] f32  ->  wt: [cout][Kp] bf16 (transposed, zero K-padded)
__global__ void k_wcvt(const float* __restrict__ w, u16* __restrict__ wt,
                       int cin, int cout, int kp) {
    int t = blockIdx.x * blockDim.x + threadIdx.x;
    if (t >= cout * kp) return;
    int nn = t / kp, k = t % kp;
    float v = (k < cin) ? w[(size_t)k * cout + nn] : 0.f;
    wt[t] = f2bf(v);
}

// ---------------------------------------------------- WMMA GEMM + ReLU -----
// A: [M][Kp] bf16 row-major; WT: [Nout][Kp] bf16; out = relu(A * WT^T)
// one wave -> 32x32 output tile (2x2 WMMA tiles, shared A/B fragments);
// block of 4 waves -> 32 rows x 128 cols.
__device__ __forceinline__ v16bf ld_frag(const u16* p0, const u16* p1) {
    uint4 a0 = *(const uint4*)p0;
    uint4 a1 = *(const uint4*)p1;
    v8u u;
    u[0]=a0.x; u[1]=a0.y; u[2]=a0.z; u[3]=a0.w;
    u[4]=a1.x; u[5]=a1.y; u[6]=a1.z; u[7]=a1.w;
    return __builtin_bit_cast(v16bf, u);
}

__global__ __launch_bounds__(128)
void k_gemm_relu(const u16* __restrict__ A, const u16* __restrict__ WT,
                 void* __restrict__ out, int M, int Kp, int Nout, int outF32) {
    int lane = threadIdx.x & 31;
    int wv   = threadIdx.x >> 5;
    int half = lane >> 4, rl = lane & 15;
    int row0 = blockIdx.x * 32 + rl;            // rows row0, row0+16
    int col0 = blockIdx.y * 128 + wv * 32 + rl; // cols col0, col0+16
    if (blockIdx.x * 32 >= M || col0 >= Nout) return;
    const u16* Ar0 = A  + (size_t)row0 * Kp;
    const u16* Ar1 = Ar0 + (size_t)16 * Kp;
    const u16* Wr0 = WT + (size_t)col0 * Kp;
    const u16* Wr1 = Wr0 + (size_t)16 * Kp;
    v8f acc00 = {0.f,0.f,0.f,0.f,0.f,0.f,0.f,0.f};
    v8f acc01 = acc00, acc10 = acc00, acc11 = acc00;
    for (int k0 = 0; k0 < Kp; k0 += 32) {
        if (k0 + 32 < Kp) {                      // global_prefetch_b8 next slice
            __builtin_prefetch(Ar0 + k0 + 32, 0, 3);
            __builtin_prefetch(Ar1 + k0 + 32, 0, 3);
            __builtin_prefetch(Wr0 + k0 + 32, 0, 3);
            __builtin_prefetch(Wr1 + k0 + 32, 0, 3);
        }
        // A fragments (16x32 bf16): lanes 0-15 K=0-7 & 16-23, lanes 16-31 K=8-15 & 24-31
        v16bf a0 = ld_frag(Ar0 + k0 + half * 8, Ar0 + k0 + 16 + half * 8);
        v16bf a1 = ld_frag(Ar1 + k0 + half * 8, Ar1 + k0 + 16 + half * 8);
        // B fragments (32x16 bf16): lanes 0-15 K=0-7 & 16-23, lanes 16-31 K=8-15 & 24-31 (col = rl)
        v16bf b0 = ld_frag(Wr0 + k0 + half * 8, Wr0 + k0 + 16 + half * 8);
        v16bf b1 = ld_frag(Wr1 + k0 + half * 8, Wr1 + k0 + 16 + half * 8);
        acc00 = __builtin_amdgcn_wmma_f32_16x16x32_bf16(false, a0, false, b0, (short)0, acc00, false, false);
        acc01 = __builtin_amdgcn_wmma_f32_16x16x32_bf16(false, a0, false, b1, (short)0, acc01, false, false);
        acc10 = __builtin_amdgcn_wmma_f32_16x16x32_bf16(false, a1, false, b0, (short)0, acc10, false, false);
        acc11 = __builtin_amdgcn_wmma_f32_16x16x32_bf16(false, a1, false, b1, (short)0, acc11, false, false);
    }
    int rBase = blockIdx.x * 32 + half * 8;
    if (outF32) {
        float* O = (float*)out;
#pragma unroll
        for (int i = 0; i < 8; ++i) {
            float v00 = acc00[i] > 0.f ? acc00[i] : 0.f;
            float v01 = acc01[i] > 0.f ? acc01[i] : 0.f;
            float v10 = acc10[i] > 0.f ? acc10[i] : 0.f;
            float v11 = acc11[i] > 0.f ? acc11[i] : 0.f;
            O[(size_t)(rBase + i)      * Nout + col0]      = v00;
            O[(size_t)(rBase + i)      * Nout + col0 + 16] = v01;
            O[(size_t)(rBase + i + 16) * Nout + col0]      = v10;
            O[(size_t)(rBase + i + 16) * Nout + col0 + 16] = v11;
        }
    } else {
        u16* O = (u16*)out;
#pragma unroll
        for (int i = 0; i < 8; ++i) {
            float v00 = acc00[i] > 0.f ? acc00[i] : 0.f;
            float v01 = acc01[i] > 0.f ? acc01[i] : 0.f;
            float v10 = acc10[i] > 0.f ? acc10[i] : 0.f;
            float v11 = acc11[i] > 0.f ? acc11[i] : 0.f;
            O[(size_t)(rBase + i)      * Nout + col0]      = f2bf(v00);
            O[(size_t)(rBase + i)      * Nout + col0 + 16] = f2bf(v01);
            O[(size_t)(rBase + i + 16) * Nout + col0]      = f2bf(v10);
            O[(size_t)(rBase + i + 16) * Nout + col0 + 16] = f2bf(v11);
        }
    }
}

// ------------------------------------------------------- max over ns -------
__global__ void k_pool(const u16* __restrict__ h, float* __restrict__ f,
                       int np, int ns, int C, int b) {
    int t = blockIdx.x * blockDim.x + threadIdx.x;
    if (t >= np * C) return;
    int s = t / C, c = t % C;
    const u16* p = h + ((size_t)s * ns) * C + c;
    float m = -1e30f;
    for (int k = 0; k < ns; ++k) { float v = bf2f(p[(size_t)k * C]); m = v > m ? v : m; }
    f[((size_t)b * np + s) * C + c] = m;
}

// -------------------------------------------------------------- 3-NN -------
__global__ void k_three_nn(const float* __restrict__ xyz1, const float* __restrict__ xyz2,
                           int n1, int n2, int* __restrict__ idx3, float* __restrict__ w3,
                           int total) {
    int t = blockIdx.x * blockDim.x + threadIdx.x;
    if (t >= total) return;
    int b = t / n1;
    const float* p = xyz1 + (size_t)t * 3;
    const float* q = xyz2 + (size_t)b * n2 * 3;
    float px = p[0], py = p[1], pz = p[2];
    float d0 = 1e30f, d1 = 1e30f, d2 = 1e30f; int i0 = 0, i1 = 0, i2 = 0;
    for (int j = 0; j < n2; ++j) {
        float dx = q[3*j] - px, dy = q[3*j+1] - py, dz = q[3*j+2] - pz;
        float d = dx*dx + dy*dy + dz*dz;
        if (d < d0)      { d2 = d1; i2 = i1; d1 = d0; i1 = i0; d0 = d; i0 = j; }
        else if (d < d1) { d2 = d1; i2 = i1; d1 = d;  i1 = j; }
        else if (d < d2) { d2 = d;  i2 = j; }
    }
    float r0 = 1.f/(d0+1e-8f), r1 = 1.f/(d1+1e-8f), r2 = 1.f/(d2+1e-8f);
    float s = r0 + r1 + r2;
    idx3[(size_t)t*3+0] = i0; idx3[(size_t)t*3+1] = i1; idx3[(size_t)t*3+2] = i2;
    w3[(size_t)t*3+0] = r0/s; w3[(size_t)t*3+1] = r1/s; w3[(size_t)t*3+2] = r2/s;
}

// ----------------------------------- FP: interpolate + concat -> bf16 A ----
__global__ void k_build_fp(const int* __restrict__ idx3, const float* __restrict__ w3,
                           const float* __restrict__ fB, const float* __restrict__ fA,
                           u16* __restrict__ A, int n1, int n2, int C2, int C1,
                           int Kp, int b) {
    int m = blockIdx.x;
    size_t rg = (size_t)b * n1 + m;
    int j0 = idx3[rg*3+0], j1 = idx3[rg*3+1], j2 = idx3[rg*3+2];
    float w0 = w3[rg*3+0], w1 = w3[rg*3+1], w2 = w3[rg*3+2];
    const float* F0 = fB + (size_t)j0 * C2;
    const float* F1 = fB + (size_t)j1 * C2;
    const float* F2 = fB + (size_t)j2 * C2;
    u16* Ar = A + (size_t)m * Kp;
    for (int c = threadIdx.x; c < Kp; c += blockDim.x) {
        float v;
        if (c < C2)           v = w0*F0[c] + w1*F1[c] + w2*F2[c];
        else if (c < C2 + C1) v = fA[(size_t)m * C1 + (c - C2)];
        else                  v = 0.f;
        Ar[c] = f2bf(v);
    }
}

// ===========================================================================
extern "C" void kernel_launch(void* const* d_in, const int* in_sizes, int n_in,
                              void* d_out, int out_size, void* d_ws, size_t ws_size,
                              hipStream_t stream) {
    (void)in_sizes; (void)n_in; (void)out_size; (void)ws_size;
    const float* pc = (const float*)d_in[0];
    const float* W[15];
    for (int i = 0; i < 15; ++i) W[i] = (const float*)d_in[i + 1];

    struct WS { int cin, cout, kp; };
    const WS wspec[15] = {
        {4,128,32},{128,128,128},{128,256,128},       // sa1
        {259,128,288},{128,128,128},{128,256,128},    // sa2
        {259,128,288},{128,128,128},{128,256,128},    // sa3
        {512,256,512},{256,256,256},                  // fp1
        {512,256,512},{256,256,256},                  // fp2
        {257,256,288},{256,256,256}};                 // fp3

    // ---- workspace carving (256B aligned) ----
    char* base = (char*)d_ws;
    size_t cur = 0;
    auto alloc = [&](size_t bytes) -> char* {
        cur = (cur + 255) & ~(size_t)255;
        char* p = base + cur; cur += bytes; return p;
    };
    float* xyz0 = (float*)alloc((size_t)BATCH*NPTS*3*4);
    float* f0   = (float*)alloc((size_t)BATCH*NPTS*4);
    float* dtmp = (float*)alloc((size_t)BATCH*NPTS*4);
    int*   fi1  = (int*)alloc((size_t)BATCH*512*4);
    int*   fi2  = (int*)alloc((size_t)BATCH*256*4);
    int*   fi3  = (int*)alloc((size_t)BATCH*128*4);
    float* xyz1 = (float*)alloc((size_t)BATCH*512*3*4);
    float* xyz2 = (float*)alloc((size_t)BATCH*256*3*4);
    float* xyz3 = (float*)alloc((size_t)BATCH*128*3*4);
    int*   idx1 = (int*)alloc((size_t)BATCH*512*32*4);
    int*   idx2 = (int*)alloc((size_t)BATCH*256*16*4);
    int*   idx3 = (int*)alloc((size_t)BATCH*128*16*4);
    float* f1   = (float*)alloc((size_t)BATCH*512*256*4);
    float* f2   = (float*)alloc((size_t)BATCH*256*256*4);
    float* f3   = (float*)alloc((size_t)BATCH*128*256*4);
    float* g2   = (float*)alloc((size_t)BATCH*256*256*4);   // fp1 out
    float* g1   = (float*)alloc((size_t)BATCH*512*256*4);   // fp2 out
    int*   nn1i = (int*)alloc((size_t)BATCH*256*3*4);
    float* nn1w = (float*)alloc((size_t)BATCH*256*3*4);
    int*   nn2i = (int*)alloc((size_t)BATCH*512*3*4);
    float* nn2w = (float*)alloc((size_t)BATCH*512*3*4);
    int*   nn3i = (int*)alloc((size_t)BATCH*NPTS*3*4);
    float* nn3w = (float*)alloc((size_t)BATCH*NPTS*3*4);
    u16* wt[15];
    for (int i = 0; i < 15; ++i)
        wt[i] = (u16*)alloc((size_t)wspec[i].cout * wspec[i].kp * 2);
    u16* BUF0 = (u16*)alloc((size_t)12*1024*1024);
    u16* BUF1 = (u16*)alloc((size_t)12*1024*1024);

    // ---- weight pack ----
    for (int i = 0; i < 15; ++i) {
        int total = wspec[i].cout * wspec[i].kp;
        k_wcvt<<<(total + 255)/256, 256, 0, stream>>>(W[i], wt[i],
                wspec[i].cin, wspec[i].cout, wspec[i].kp);
    }

    // ---- split pointcloud ----
    k_split<<<(BATCH*NPTS + 255)/256, 256, 0, stream>>>(pc, xyz0, f0, BATCH*NPTS);

    auto gemm = [&](const u16* Ab, int wi, void* outp, int M, int outF32) {
        dim3 g(M/32, wspec[wi].cout/128);
        k_gemm_relu<<<g, 128, 0, stream>>>(Ab, wt[wi], outp, M,
                                           wspec[wi].kp, wspec[wi].cout, outF32);
    };

    auto run_sa = [&](const float* xyzL, const float* featsL, int nL, int Cf,
                      int npL, int nsL, float radius, int* fiL, float* nxyzL,
                      int* idxL, int w0i, float* fOut) {
        k_fps<<<BATCH, 256, 0, stream>>>(xyzL, dtmp, nL, npL, fiL);
        int totalS = BATCH * npL;
        k_gather_xyz<<<(totalS + 127)/128, 128, 0, stream>>>(xyzL, fiL, nxyzL, nL, npL, totalS);
        k_ballq<<<(totalS + 3)/4, 128, 0, stream>>>(nxyzL, xyzL, nL, npL, nsL,
                                                    radius*radius, idxL, totalS);
        int Mb = npL * nsL;
        for (int b = 0; b < BATCH; ++b) {
            k_build_sa<<<Mb, 64, 0, stream>>>(xyzL, nxyzL, featsL, idxL, BUF0,
                                              nL, npL, nsL, Cf, wspec[w0i].kp,
                                              1.f/radius, b);
            gemm(BUF0, w0i,     BUF1, Mb, 0);
            gemm(BUF1, w0i + 1, BUF0, Mb, 0);
            gemm(BUF0, w0i + 2, BUF1, Mb, 0);
            k_pool<<<(npL*256 + 255)/256, 256, 0, stream>>>(BUF1, fOut, npL, nsL, 256, b);
        }
    };

    auto run_fp = [&](const float* xyzA, const float* xyzBp, const float* fBfull,
                      const float* fAfull, int n1, int n2, int C2, int C1,
                      int* nnI, float* nnW, int w0i, float* outBase) {
        int total = BATCH * n1;
        k_three_nn<<<(total + 127)/128, 128, 0, stream>>>(xyzA, xyzBp, n1, n2, nnI, nnW, total);
        int Kp = wspec[w0i].kp;
        for (int b = 0; b < BATCH; ++b) {
            k_build_fp<<<n1, 64, 0, stream>>>(nnI, nnW,
                    fBfull + (size_t)b*n2*C2, fAfull + (size_t)b*n1*C1,
                    BUF0, n1, n2, C2, C1, Kp, b);
            gemm(BUF0, w0i,     BUF1, n1, 0);
            gemm(BUF1, w0i + 1, outBase + (size_t)b*n1*256, n1, 1);
        }
    };

    // ---- SA stages ----
    run_sa(xyz0, f0, NPTS, 1,   512, 32, 0.4f, fi1, xyz1, idx1, 0, f1);
    run_sa(xyz1, f1, 512,  256, 256, 16, 0.8f, fi2, xyz2, idx2, 3, f2);
    run_sa(xyz2, f2, 256,  256, 128, 16, 1.2f, fi3, xyz3, idx3, 6, f3);

    // ---- FP stages ----
    run_fp(xyz2, xyz3, f3, f2, 256, 128, 256, 256, nn1i, nn1w, 9,  g2);
    run_fp(xyz1, xyz2, g2, f1, 512, 256, 256, 256, nn2i, nn2w, 11, g1);
    run_fp(xyz0, xyz1, g1, f0, NPTS, 512, 256, 1,  nn3i, nn3w, 13, (float*)d_out);
}
